// WeightOnlyPerChannelQuantizedLinear_60430189855203
// MI455X (gfx1250) — 8-bit matrix operand order repaired and checked by reading, compiled, not run on hardware
//
#include <hip/hip_runtime.h>
#include <hip/hip_bf16.h>
#include <stdint.h>

typedef int v8i __attribute__((ext_vector_type(8)));

#define MM 512
#define KK 8192
#define NN 28672
#define TILE_K 128
#define BLOCK_M 128
#define BLOCK_N 64
#define A_LDS_STRIDE 144
#define A_TILE_BYTES (BLOCK_M * A_LDS_STRIDE)

__device__ __forceinline__ void async_copy_b128(unsigned lds_off, const void* gaddr) {
    unsigned long long ga = (unsigned long long)(uintptr_t)gaddr;
    asm volatile("global_load_async_to_lds_b128 %0, %1, off"
                 :: "v"(lds_off), "v"(ga) : "memory");
}
__device__ __forceinline__ void wait_async0() {
    asm volatile("s_wait_asynccnt 0" ::: "memory");
}

__global__ void wq_quant_rows_kernel(const float* __restrict__ in,
                                     int8_t* __restrict__ q,
                                     float* __restrict__ act_s) {
    const int row = blockIdx.x;
    const int t   = threadIdx.x;
    const float4* in4 = (const float4*)(in + (size_t)row * KK);
    float4 v[8];
    float amax = 0.f;
#pragma unroll
    for (int i = 0; i < 8; ++i) {
        v[i] = in4[i * 256 + t];
        amax = fmaxf(amax, fmaxf(fmaxf(fabsf(v[i].x), fabsf(v[i].y)),
                                 fmaxf(fabsf(v[i].z), fabsf(v[i].w))));
    }
#pragma unroll
    for (int off = 16; off > 0; off >>= 1)
        amax = fmaxf(amax, __shfl_xor(amax, off, 32));
    __shared__ float red[8];
    const int wave = t >> 5, lane = t & 31;
    if (lane == 0) red[wave] = amax;
    __syncthreads();
    if (wave == 0) {
        float m = (lane < 8) ? red[lane] : 0.f;
#pragma unroll
        for (int off = 4; off > 0; off >>= 1)
            m = fmaxf(m, __shfl_xor(m, off, 32));
        if (lane == 0) red[0] = fmaxf(m, 1e-5f);
    }
    __syncthreads();
    const float mx  = red[0];
    const float inv = 127.0f / mx;
    if (t == 0) act_s[row] = mx / 127.0f;

    uint32_t* q4 = (uint32_t*)(q + (size_t)row * KK);
#pragma unroll
    for (int i = 0; i < 8; ++i) {
        int a = (int)__builtin_rintf(v[i].x * inv);
        int b = (int)__builtin_rintf(v[i].y * inv);
        int c = (int)__builtin_rintf(v[i].z * inv);
        int d = (int)__builtin_rintf(v[i].w * inv);
        a = a > 127 ? 127 : (a < -128 ? -128 : a);
        b = b > 127 ? 127 : (b < -128 ? -128 : b);
        c = c > 127 ? 127 : (c < -128 ? -128 : c);
        d = d > 127 ? 127 : (d < -128 ? -128 : d);
        q4[i * 256 + t] = (uint32_t)(a & 0xFF) | ((uint32_t)(b & 0xFF) << 8) |
                          ((uint32_t)(c & 0xFF) << 16) | ((uint32_t)(d & 0xFF) << 24);
    }
}

struct BFrag { uint4 lo, hi; };

__device__ __forceinline__ BFrag load_bfrag(const int8_t* wp) {
    BFrag f;
    f.lo = *(const uint4*)(wp);
    f.hi = *(const uint4*)(wp + 32);
    return f;
}

__device__ __forceinline__ v8i load_afrag(const unsigned char* ar) {
    const uint2 d0 = *(const uint2*)(ar);
    const uint2 d1 = *(const uint2*)(ar + 8);
    const uint2 d2 = *(const uint2*)(ar + 32);
    const uint2 d3 = *(const uint2*)(ar + 40);
    v8i a;
    a[0] = (int)d0.x; a[1] = (int)d0.y; a[2] = (int)d1.x; a[3] = (int)d1.y;
    a[4] = (int)d2.x; a[5] = (int)d2.y; a[6] = (int)d3.x; a[7] = (int)d3.y;
    return a;
}

__device__ __forceinline__ void compute_chunk(const unsigned char* __restrict__ abase,
                                              int rowBase, int half,
                                              const BFrag& bf, v8i acc[4]) {
    v8i bfrag;
    bfrag[0] = (int)bf.lo.x; bfrag[1] = (int)bf.lo.y; bfrag[2] = (int)bf.lo.z; bfrag[3] = (int)bf.lo.w;
    bfrag[4] = (int)bf.hi.x; bfrag[5] = (int)bf.hi.y; bfrag[6] = (int)bf.hi.z; bfrag[7] = (int)bf.hi.w;
    const unsigned char* ar = abase + rowBase * A_LDS_STRIDE + half * 16;

    v8i a0 = load_afrag(ar);
    v8i a1 = load_afrag(ar + 16 * A_LDS_STRIDE);
    acc[0] = __builtin_amdgcn_wmma_i32_16x16x64_iu8(true, a0, true, bfrag, acc[0], false, false);
    v8i a2 = load_afrag(ar + 32 * A_LDS_STRIDE);
    acc[1] = __builtin_amdgcn_wmma_i32_16x16x64_iu8(true, a1, true, bfrag, acc[1], false, false);
    v8i a3 = load_afrag(ar + 48 * A_LDS_STRIDE);
    acc[2] = __builtin_amdgcn_wmma_i32_16x16x64_iu8(true, a2, true, bfrag, acc[2], false, false);
    acc[3] = __builtin_amdgcn_wmma_i32_16x16x64_iu8(true, a3, true, bfrag, acc[3], false, false);
}

__global__ void __launch_bounds__(256)
wq_gemm_i8_wmma_kernel(const int8_t* __restrict__ q,
                       const int8_t* __restrict__ weight,
                       const __hip_bfloat16* __restrict__ wscale,
                       const float* __restrict__ act_s,
                       float* __restrict__ out) {
    __shared__ __align__(16) unsigned char atile[2 * A_TILE_BYTES];

    const int t    = threadIdx.x;
    const int wave = t >> 5;
    const int lane = t & 31;
    const int wm   = wave & 1;
    const int wn   = wave >> 1;
    const int half = lane >> 4;
    const int l16  = lane & 15;

    const int nBlock = blockIdx.x * BLOCK_N;
    const int mBlock = blockIdx.y * BLOCK_M;

    const int srow = t >> 3;
    const int scol = (t & 7) * 16;
    const int8_t* aGlobBase = q + (size_t)mBlock * KK + scol;
    const unsigned ldsAbs = (unsigned)(uintptr_t)&atile[0];
    size_t   gOff[4];
    unsigned lOff[4];
#pragma unroll
    for (int r = 0; r < 4; ++r) {
        gOff[r] = (size_t)(srow + r * 32) * KK;
        lOff[r] = ldsAbs + (srow + r * 32) * A_LDS_STRIDE + scol;
    }

    const int nCol = nBlock + wn * 16 + l16;
    const int8_t* wRow = weight + (size_t)nCol * KK + half * 16;

    const unsigned char* abase0 = &atile[0];
    const unsigned char* abase1 = &atile[A_TILE_BYTES];
    const int rowBase = wm * 64 + l16;

    v8i acc[4] = {};

#pragma unroll
    for (int r = 0; r < 4; ++r)
        async_copy_b128(lOff[r], aGlobBase + gOff[r]);
    BFrag bA = load_bfrag(wRow);
    wait_async0();
    __syncthreads();

    const int NKT = KK / TILE_K;
    int mt = 0;
    for (; mt < NKT - 2; mt += 2) {
        const int k0 = mt * TILE_K;
        {
#pragma unroll
            for (int r = 0; r < 4; ++r)
                async_copy_b128(lOff[r] + A_TILE_BYTES, aGlobBase + gOff[r] + (k0 + TILE_K));
            __builtin_prefetch(wRow + k0 + 2048, 0, 1);

            BFrag bB = load_bfrag(wRow + k0 + 64);
            compute_chunk(abase0,      rowBase, half, bA, acc);
            bA = load_bfrag(wRow + k0 + TILE_K);
            compute_chunk(abase0 + 64, rowBase, half, bB, acc);

            wait_async0();
            __syncthreads();
        }
        {
            const int k1 = k0 + TILE_K;
#pragma unroll
            for (int r = 0; r < 4; ++r)
                async_copy_b128(lOff[r], aGlobBase + gOff[r] + (k1 + TILE_K));
            __builtin_prefetch(wRow + k1 + 2048, 0, 1);

            BFrag bB = load_bfrag(wRow + k1 + 64);
            compute_chunk(abase1,      rowBase, half, bA, acc);
            bA = load_bfrag(wRow + k1 + TILE_K);
            compute_chunk(abase1 + 64, rowBase, half, bB, acc);

            wait_async0();
            __syncthreads();
        }
    }
    {
        const int k0 = mt * TILE_K;
#pragma unroll
        for (int r = 0; r < 4; ++r)
            async_copy_b128(lOff[r] + A_TILE_BYTES, aGlobBase + gOff[r] + (k0 + TILE_K));

        BFrag bB = load_bfrag(wRow + k0 + 64);
        compute_chunk(abase0,      rowBase, half, bA, acc);
        bA = load_bfrag(wRow + k0 + TILE_K);
        compute_chunk(abase0 + 64, rowBase, half, bB, acc);

        wait_async0();
        __syncthreads();

        const int k1 = k0 + TILE_K;
        bB = load_bfrag(wRow + k1 + 64);
        compute_chunk(abase1,      rowBase, half, bA, acc);
        compute_chunk(abase1 + 64, rowBase, half, bB, acc);
    }

    const float ws = __bfloat162float(wscale[nCol]);
#pragma unroll
    for (int i = 0; i < 4; ++i) {
        const int mBase = mBlock + wm * 64 + i * 16 + half * 8;
#pragma unroll
        for (int j = 0; j < 8; ++j) {
            const int m = mBase + j;
            const float av   = __bfloat162float(__float2bfloat16((float)acc[i][j]));
            const float prod = __bfloat162float(__float2bfloat16(av * ws));
            out[(size_t)m * NN + nCol] = prod * act_s[m];
        }
    }
}

extern "C" void kernel_launch(void* const* d_in, const int* in_sizes, int n_in,
                              void* d_out, int out_size, void* d_ws, size_t ws_size,
                              hipStream_t stream) {
    (void)in_sizes; (void)n_in; (void)out_size; (void)ws_size;
    const float*           inputs = (const float*)d_in[0];
    const int8_t*          weight = (const int8_t*)d_in[1];
    const __hip_bfloat16*  wsc    = (const __hip_bfloat16*)d_in[2];
    float*                 out    = (float*)d_out;

    int8_t* qbuf = (int8_t*)d_ws;
    float*  acts = (float*)((char*)d_ws + (size_t)MM * KK);

    wq_quant_rows_kernel<<<MM, 256, 0, stream>>>(inputs, qbuf, acts);

    dim3 grid(NN / BLOCK_N, MM / BLOCK_M);
    wq_gemm_i8_wmma_kernel<<<grid, 256, 0, stream>>>(qbuf, weight, wsc, acts, out);
}
